// Head_78881369358382
// MI455X (gfx1250) — matrix operand order repaired and checked by reading, not compiled, not run on hardware
//
#include <hip/hip_runtime.h>
#include <hip/hip_bf16.h>
#include <math.h>

typedef __bf16 bf16;
typedef __attribute__((ext_vector_type(16))) __bf16 v16bf;
typedef __attribute__((ext_vector_type(8)))  __bf16 bf16x8;
typedef __attribute__((ext_vector_type(8)))  float  v8f;

#define B_ 128
#define T_ 256
#define C_ 1024
#define H_ 128

// B-operand (32x16 bf16): lane covers K = 8*kh+0..7 and 16+8*kh+0..7.
__device__ __forceinline__ v16bf load_b_frag(const bf16* row, int kh) {
    bf16x8 lo = *(const bf16x8*)(row + 8 * kh);
    bf16x8 hi = *(const bf16x8*)(row + 16 + 8 * kh);
    v16bf r;
    #pragma unroll
    for (int i = 0; i < 8; ++i) { r[i] = lo[i]; r[8 + i] = hi[i]; }
    return r;
}
// A-operand (16x32 bf16): lane covers K = 8*kh+0..7 and 16+8*kh+0..7.
__device__ __forceinline__ v16bf load_a_frag(const bf16* row, int kh) {
    bf16x8 lo = *(const bf16x8*)(row + 8 * kh);
    bf16x8 hi = *(const bf16x8*)(row + 16 + 8 * kh);
    v16bf r;
    #pragma unroll
    for (int i = 0; i < 8; ++i) { r[i] = lo[i]; r[8 + i] = hi[i]; }
    return r;
}

// ---------------------------------------------------------------------------
// Kernel 0: one-time weight convert+transpose: W[C,H] f32 -> WT[3][H][C] bf16
// ---------------------------------------------------------------------------
__global__ void wprep_kernel(const float* __restrict__ Wk,
                             const float* __restrict__ Wq,
                             const float* __restrict__ Wv,
                             bf16* __restrict__ wt)
{
    const int w = blockIdx.x >> 7;            // 0:q 1:k 2:v
    const int h = blockIdx.x & 127;
    const float* W = (w == 0) ? Wq : (w == 1) ? Wk : Wv;
    bf16* dst = wt + ((size_t)w * H_ + h) * C_;
    for (int c = threadIdx.x; c < C_; c += 256)
        dst[c] = (bf16)W[(size_t)c * H_ + h];
}

// ---------------------------------------------------------------------------
// Kernel 1: fused q/k/v projection. One wave = 16 rows x 64 H-cols (4 tiles)
// for all three weights (x A-fragment loaded once, 12 WMMAs per K-chunk).
// ---------------------------------------------------------------------------
__global__ __launch_bounds__(256) void proj_kernel(
    const float* __restrict__ x, const bf16* __restrict__ wt,
    bf16* __restrict__ qb, bf16* __restrict__ kb, bf16* __restrict__ vtb)
{
    const int lane = threadIdx.x & 31;
    const int wave = threadIdx.x >> 5;
    const int job = blockIdx.x * 8 + wave;    // 0..4095
    const int strip = job >> 1;
    const int t0 = (job & 1) * 4;             // first of 4 H-tiles
    const int m0 = strip * 16;                // row in [B*T]
    const int n  = lane & 15;
    const int kh = lane >> 4;

    v8f acc[3][4] = {};
    const float* xrow = x + (size_t)(m0 + n) * C_;
    for (int cc = 0; cc < C_; cc += 32) {
        v16bf a;
        {
            float4 f0 = *(const float4*)(xrow + cc + 8 * kh);
            float4 f1 = *(const float4*)(xrow + cc + 8 * kh + 4);
            float4 f2 = *(const float4*)(xrow + cc + 16 + 8 * kh);
            float4 f3 = *(const float4*)(xrow + cc + 16 + 8 * kh + 4);
            a[0]=(bf16)f0.x;  a[1]=(bf16)f0.y;  a[2]=(bf16)f0.z;  a[3]=(bf16)f0.w;
            a[4]=(bf16)f1.x;  a[5]=(bf16)f1.y;  a[6]=(bf16)f1.z;  a[7]=(bf16)f1.w;
            a[8]=(bf16)f2.x;  a[9]=(bf16)f2.y;  a[10]=(bf16)f2.z; a[11]=(bf16)f2.w;
            a[12]=(bf16)f3.x; a[13]=(bf16)f3.y; a[14]=(bf16)f3.z; a[15]=(bf16)f3.w;
        }
        __builtin_prefetch(xrow + cc + 64, 0, 0);
        #pragma unroll
        for (int w = 0; w < 3; ++w)
            #pragma unroll
            for (int t = 0; t < 4; ++t) {
                const bf16* wrow =
                    wt + ((size_t)w * H_ + (t0 + t) * 16 + n) * C_ + cc;
                v16bf bfr = load_b_frag(wrow, kh);
                acc[w][t] = __builtin_amdgcn_wmma_f32_16x16x32_bf16(
                    false, a, false, bfr, (short)0, acc[w][t], false, false);
            }
    }
    #pragma unroll
    for (int w = 0; w < 2; ++w) {
        bf16* dst = (w == 0) ? qb : kb;
        #pragma unroll
        for (int t = 0; t < 4; ++t)
            #pragma unroll
            for (int r = 0; r < 8; ++r)
                dst[(size_t)(m0 + 8 * kh + r) * H_ + (t0 + t) * 16 + n] =
                    (bf16)acc[w][t][r];
    }
    const int b  = m0 / T_;
    const int tt = (m0 % T_) + 8 * kh;
    #pragma unroll
    for (int t = 0; t < 4; ++t) {
        union { uint4 u4; bf16 h[8]; } pk;
        #pragma unroll
        for (int r = 0; r < 8; ++r) pk.h[r] = (bf16)acc[2][t][r];
        *(uint4*)(vtb + ((size_t)b * H_ + (t0 + t) * 16 + n) * T_ + tt) = pk.u4;
    }
}

// ---------------------------------------------------------------------------
// Kernel 2: causal flash attention. Block = 8 query strips of ONE batch; the
// 32-key K/V chunks are shared by all waves -> staged once per block in LDS
// via async global->LDS loads, double-buffered (ASYNCcnt), WMMA on top.
// ---------------------------------------------------------------------------
__global__ __launch_bounds__(256) void attn_kernel(
    const bf16* __restrict__ qb, const bf16* __restrict__ kb,
    const bf16* __restrict__ vtb, float* __restrict__ out)
{
    __shared__ bf16 kbuf[2][32][136];   // [buf][key][channel], padded rows
    __shared__ bf16 vbuf[2][128][40];   // [buf][h][key], padded rows
    __shared__ bf16 plds[8][16][40];    // per-wave P transpose staging

    const int tid  = threadIdx.x;
    const int lane = tid & 31;
    const int wave = tid >> 5;
    const int b  = blockIdx.x >> 1;
    const int qt = (blockIdx.x & 1) * 8 + wave;   // query strip 0..15
    const int n  = lane & 15;
    const int kh = lane >> 4;
    const float scale = 0.03125f;                 // 1024^-0.5

    // stage one 32-key chunk of K and V into LDS buffer `buf` (async)
    auto stage = [&](int s0, int buf) {
        #pragma unroll
        for (int i = 0; i < 2; ++i) {             // K: 32 rows x 16 segs
            int task = tid * 2 + i;
            int row = task >> 4, seg = task & 15;
            const bf16* src = kb + ((size_t)b * T_ + s0 + row) * H_ + seg * 8;
            unsigned dst = (unsigned)(uintptr_t)&kbuf[buf][row][seg * 8];
            asm volatile("global_load_async_to_lds_b128 %0, %1, off"
                         :: "v"(dst), "v"(src) : "memory");
        }
        #pragma unroll
        for (int i = 0; i < 2; ++i) {             // V: 128 rows x 4 segs
            int task = tid * 2 + i;
            int h = task >> 2, seg = task & 3;
            const bf16* src = vtb + ((size_t)b * H_ + h) * T_ + s0 + seg * 8;
            unsigned dst = (unsigned)(uintptr_t)&vbuf[buf][h][seg * 8];
            asm volatile("global_load_async_to_lds_b128 %0, %1, off"
                         :: "v"(dst), "v"(src) : "memory");
        }
    };

    v16bf qf[4];
    const bf16* qrow = qb + ((size_t)b * T_ + qt * 16 + n) * H_;
    #pragma unroll
    for (int cc = 0; cc < 4; ++cc) qf[cc] = load_a_frag(qrow + cc * 32, kh);

    v8f o[8] = {};
    float mrow[8], lrow[8];
    #pragma unroll
    for (int r = 0; r < 8; ++r) { mrow[r] = -INFINITY; lrow[r] = 0.0f; }

    const int mychunks  = qt / 2 + 1;
    const int blkchunks = (((blockIdx.x & 1) ? 15 : 7) / 2) + 1;

    stage(0, 0);
    for (int kc = 0; kc < blkchunks; ++kc) {
        const int buf = kc & 1;
        const int s0  = kc * 32;
        if (kc + 1 < blkchunks) {
            stage(s0 + 32, buf ^ 1);
            asm volatile("s_wait_asynccnt 0x4" ::: "memory");
        } else {
            asm volatile("s_wait_asynccnt 0x0" ::: "memory");
        }
        __syncthreads();                 // chunk kc resident for all waves

        if (kc < mychunks) {             // wave-uniform guard (EXEC stays full)
            v8f s[2] = {};
            #pragma unroll
            for (int tl = 0; tl < 2; ++tl)
                #pragma unroll
                for (int cc = 0; cc < 4; ++cc) {
                    v16bf bfr = load_b_frag(&kbuf[buf][tl * 16 + n][cc * 32], kh);
                    s[tl] = __builtin_amdgcn_wmma_f32_16x16x32_bf16(
                        false, qf[cc], false, bfr, (short)0, s[tl], false, false);
                }
            #pragma unroll
            for (int r = 0; r < 8; ++r) {
                const int row = qt * 16 + 8 * kh + r;
                float v0 = s[0][r] * scale; if (s0 + n      > row) v0 = -INFINITY;
                float v1 = s[1][r] * scale; if (s0 + 16 + n > row) v1 = -INFINITY;
                float mx = fmaxf(v0, v1);
                mx = fmaxf(mx, __shfl_xor(mx, 1));
                mx = fmaxf(mx, __shfl_xor(mx, 2));
                mx = fmaxf(mx, __shfl_xor(mx, 4));
                mx = fmaxf(mx, __shfl_xor(mx, 8));
                const float mnew = fmaxf(mrow[r], mx);
                const float p0 = __expf(v0 - mnew);
                const float p1 = __expf(v1 - mnew);
                float rs = p0 + p1;
                rs += __shfl_xor(rs, 1);
                rs += __shfl_xor(rs, 2);
                rs += __shfl_xor(rs, 4);
                rs += __shfl_xor(rs, 8);
                const float alpha = __expf(mrow[r] - mnew);
                lrow[r] = lrow[r] * alpha + rs;
                mrow[r] = mnew;
                #pragma unroll
                for (int t = 0; t < 8; ++t) o[t][r] *= alpha;
                plds[wave][8 * kh + r][n]      = (bf16)p0;
                plds[wave][8 * kh + r][16 + n] = (bf16)p1;
            }
            v16bf pa;                    // P: D layout -> A layout (wave-private)
            {
                bf16x8 lo = *(const bf16x8*)&plds[wave][n][8 * kh];
                bf16x8 hi = *(const bf16x8*)&plds[wave][n][16 + 8 * kh];
                #pragma unroll
                for (int i = 0; i < 8; ++i) { pa[i] = lo[i]; pa[8 + i] = hi[i]; }
            }
            #pragma unroll
            for (int t = 0; t < 8; ++t) {
                v16bf bfr = load_b_frag(&vbuf[buf][t * 16 + n][0], kh);
                o[t] = __builtin_amdgcn_wmma_f32_16x16x32_bf16(
                    false, pa, false, bfr, (short)0, o[t], false, false);
            }
        }
        asm volatile("s_wait_dscnt 0x0" ::: "memory");
        __syncthreads();                 // reads done before buffer reuse
    }

    float inv[8];
    #pragma unroll
    for (int r = 0; r < 8; ++r) inv[r] = 1.0f / lrow[r];
    #pragma unroll
    for (int t = 0; t < 8; ++t)
        #pragma unroll
        for (int r = 0; r < 8; ++r)
            out[((size_t)b * T_ + qt * 16 + 8 * kh + r) * H_ + t * 16 + n] =
                o[t][r] * inv[r];
}

extern "C" void kernel_launch(void* const* d_in, const int* in_sizes, int n_in,
                              void* d_out, int out_size, void* d_ws, size_t ws_size,
                              hipStream_t stream) {
    const float* x  = (const float*)d_in[0];
    const float* Wk = (const float*)d_in[1];
    const float* Wq = (const float*)d_in[2];
    const float* Wv = (const float*)d_in[3];
    float* out = (float*)d_out;

    const size_t nqkv = (size_t)B_ * T_ * H_;
    bf16* qb  = (bf16*)d_ws;                       // [B,T,H]
    bf16* kb  = qb + nqkv;                         // [B,T,H]
    bf16* vtb = kb + nqkv;                         // [B,H,T]
    bf16* wt  = vtb + nqkv;                        // [3,H,C]

    wprep_kernel<<<384, 256, 0, stream>>>(Wk, Wq, Wv, wt);
    proj_kernel<<<512, 256, 0, stream>>>(x, wt, qb, kb, vtb);
    attn_kernel<<<256, 256, 0, stream>>>(qb, kb, vtb, out);
}
